// NSM_Conv_10926396801043
// MI455X (gfx1250) — compile-verified, matrix operand order checked by reading, not run on hardware
//
#include <hip/hip_runtime.h>

// NSM lateral-inhibition network, fused conv + 40-iteration relu recurrence.
// One workgroup (256 threads = 8 wave32) per (batch, out_row): 121 pixels
// padded to 128 columns. Channel dim = 128.
// Per-wave blocking: 2 row-tiles x 4 col-tiles (32x64 output block) so each
// B fragment read from LDS is reused by 2 WMMAs -> LDS traffic halved vs 1x8.
// All GEMM work on v_wmma_f32_16x16x32_f16; y lives in LDS as f16.

typedef __attribute__((ext_vector_type(16))) _Float16 v16h;
typedef __attribute__((ext_vector_type(8)))  float    v8f;

#define SP 200   // im2col LDS stride (halves), 192 K + pad, 16B-aligned rows
#define SY 136   // Y LDS stride (halves), 128 ch + pad, row stride 272B

union FragU { float4 f[2]; v16h h; };

// B fragment (32Kx16N, f16) from pixel-major LDS [p][k]:
// lane l: n = l&15, hi = l>>4 ; elems 0..7 = K k0..k0+7, elems 8..15 = K k0+16..k0+23
__device__ __forceinline__ v16h load_fragB(const _Float16* sm, int stride,
                                           int ct, int kc, int lane) {
    int n = lane & 15, hi = lane >> 4;
    int p  = ct * 16 + n;
    int k0 = kc * 32 + hi * 8;
    const _Float16* base = sm + p * stride + k0;
    FragU u;
    u.f[0] = *reinterpret_cast<const float4*>(base);
    u.f[1] = *reinterpret_cast<const float4*>(base + 16);
    return u.h;
}

// A fragment (16Mx32K) straight from a row-major f32 matrix in global memory,
// converted to f16. Same element<->K mapping as B.
__device__ __forceinline__ v16h load_fragA_w(const float* __restrict__ W, int ldk,
                                             int rt, int kc, int lane) {
    int m  = rt * 16 + (lane & 15);
    int hi = lane >> 4;
    int k0 = kc * 32 + hi * 8;
    const float* row = W + (size_t)m * ldk + k0;
    float g[16];
    *reinterpret_cast<float4*>(&g[0])  = *reinterpret_cast<const float4*>(row);
    *reinterpret_cast<float4*>(&g[4])  = *reinterpret_cast<const float4*>(row + 4);
    *reinterpret_cast<float4*>(&g[8])  = *reinterpret_cast<const float4*>(row + 16);
    *reinterpret_cast<float4*>(&g[12]) = *reinterpret_cast<const float4*>(row + 20);
    v16h h;
#pragma unroll
    for (int e = 0; e < 8; ++e) h[e]     = (_Float16)g[e];
#pragma unroll
    for (int e = 0; e < 8; ++e) h[8 + e] = (_Float16)g[8 + e];
    return h;
}

// A fragment of  -(comp_w - I)  = I - comp_w   (so D = A*Y + Wx gives Wx - M*y)
__device__ __forceinline__ v16h load_fragA_negM(const float* __restrict__ W,
                                                int rt, int kc, int lane) {
    int m  = rt * 16 + (lane & 15);
    int hi = lane >> 4;
    int k0 = kc * 32 + hi * 8;
    const float* row = W + (size_t)m * 128 + k0;
    float g[16];
    *reinterpret_cast<float4*>(&g[0])  = *reinterpret_cast<const float4*>(row);
    *reinterpret_cast<float4*>(&g[4])  = *reinterpret_cast<const float4*>(row + 4);
    *reinterpret_cast<float4*>(&g[8])  = *reinterpret_cast<const float4*>(row + 16);
    *reinterpret_cast<float4*>(&g[12]) = *reinterpret_cast<const float4*>(row + 20);
    v16h h;
#pragma unroll
    for (int e = 0; e < 8; ++e) {
        int j = k0 + e;
        h[e] = (_Float16)(((j == m) ? 1.0f : 0.0f) - g[e]);
    }
#pragma unroll
    for (int e = 0; e < 8; ++e) {
        int j = k0 + 16 + e;
        h[8 + e] = (_Float16)(((j == m) ? 1.0f : 0.0f) - g[8 + e]);
    }
    return h;
}

__global__ __launch_bounds__(256)
void NSM_Conv_10926396801043_kernel(const float* __restrict__ x,
                                    const float* __restrict__ enc_w,
                                    const float* __restrict__ enc_b,
                                    const float* __restrict__ comp_w,
                                    float* __restrict__ out) {
    __shared__ _Float16 sm[25600];          // 51.2 KB: im2col (128xSP) then Y (128xSY)

    const int tid  = threadIdx.x;
    const int lane = tid & 31;
    const int wv   = tid >> 5;
    const int rb   = wv >> 1;               // row-block 0..3  (rows 32*rb..+31)
    const int cb   = wv & 1;                // col-block 0..1  (pixels 64*cb..+63)
    const int hi   = lane >> 4;
    const int n    = lane & 15;
    const int b    = blockIdx.x / 121;
    const int r    = blockIdx.x % 121;

    // ---- Phase 1: im2col patch -> LDS (pixel-major [p][k], f16) ------------
    const float* xb = x + (size_t)b * 3 * 128 * 128;
    for (int j = tid; j < 128 * 192; j += 256) {
        int p = j / 192, k = j - p * 192;
        int c = k >> 6, rem = k & 63, dr = rem >> 3, dc = rem & 7;
        float v = 0.0f;
        if (p < 121) v = xb[((size_t)c * 128 + (r + dr)) * 128 + (p + dc)];
        sm[p * SP + k] = (_Float16)v;
    }
    __syncthreads();

    // ---- Phase 2: conv as 128x192 @ 192x128 WMMA GEMM, bias in C ----------
    v8f wx[2][4];
#pragma unroll
    for (int rtl = 0; rtl < 2; ++rtl) {
        const float* bp = enc_b + (rb * 2 + rtl) * 16 + hi * 8;
        float4 b0 = *reinterpret_cast<const float4*>(bp);
        float4 b1 = *reinterpret_cast<const float4*>(bp + 4);
        v8f bias = (v8f){b0.x, b0.y, b0.z, b0.w, b1.x, b1.y, b1.z, b1.w};
#pragma unroll
        for (int c = 0; c < 4; ++c) wx[rtl][c] = bias;
    }
#pragma unroll
    for (int kc = 0; kc < 6; ++kc) {
        v16h a0 = load_fragA_w(enc_w, 192, rb * 2 + 0, kc, lane);
        v16h a1 = load_fragA_w(enc_w, 192, rb * 2 + 1, kc, lane);
#pragma unroll
        for (int c = 0; c < 4; ++c) {
            v16h bf = load_fragB(sm, SP, cb * 4 + c, kc, lane);
            wx[0][c] = __builtin_amdgcn_wmma_f32_16x16x32_f16(
                false, a0, false, bf, (short)0, wx[0][c], false, false);
            wx[1][c] = __builtin_amdgcn_wmma_f32_16x16x32_f16(
                false, a1, false, bf, (short)0, wx[1][c], false, false);
        }
    }
    __syncthreads();                        // im2col reads complete

    // ---- Phase 3: hold A = I - comp_w in VGPRs; zero Y in LDS -------------
    v16h mA[2][4];
#pragma unroll
    for (int rtl = 0; rtl < 2; ++rtl)
#pragma unroll
        for (int kc = 0; kc < 4; ++kc)
            mA[rtl][kc] = load_fragA_negM(comp_w, rb * 2 + rtl, kc, lane);
    {
        int* smi = reinterpret_cast<int*>(sm);
        for (int j = tid; j < (128 * SY) / 2; j += 256) smi[j] = 0;
    }
    __syncthreads();

    // ---- Phase 4: 40 iterations  y <- relu(Wx - M y)  ---------------------
    for (int t = 0; t < 40; ++t) {
        v8f acc[2][4];
#pragma unroll
        for (int rtl = 0; rtl < 2; ++rtl)
#pragma unroll
            for (int c = 0; c < 4; ++c) acc[rtl][c] = wx[rtl][c];
#pragma unroll
        for (int kc = 0; kc < 4; ++kc) {
#pragma unroll
            for (int c = 0; c < 4; ++c) {  // 8 independent chains per kc
                v16h bf = load_fragB(sm, SY, cb * 4 + c, kc, lane);
                acc[0][c] = __builtin_amdgcn_wmma_f32_16x16x32_f16(
                    false, mA[0][kc], false, bf, (short)0, acc[0][c], false, false);
                acc[1][c] = __builtin_amdgcn_wmma_f32_16x16x32_f16(
                    false, mA[1][kc], false, bf, (short)0, acc[1][c], false, false);
            }
        }
        __syncthreads();                    // all waves done reading Y
        if (t < 39) {
#pragma unroll
            for (int rtl = 0; rtl < 2; ++rtl) {
#pragma unroll
                for (int c = 0; c < 4; ++c) {
                    union { int4 i4; _Float16 h[8]; } u;
#pragma unroll
                    for (int v = 0; v < 8; ++v)
                        u.h[v] = (_Float16)fmaxf(acc[rtl][c][v], 0.0f);
                    int p   = (cb * 4 + c) * 16 + n;
                    int ch0 = (rb * 2 + rtl) * 16 + hi * 8;
                    *reinterpret_cast<int4*>(&sm[p * SY + ch0]) = u.i4;
                }
            }
            __syncthreads();                // Y update visible to next iter
        } else {
            // final iteration: write full-precision f32 result, mask pad cols
#pragma unroll
            for (int rtl = 0; rtl < 2; ++rtl) {
#pragma unroll
                for (int c = 0; c < 4; ++c) {
                    int p = (cb * 4 + c) * 16 + n;
                    if (p < 121) {
#pragma unroll
                        for (int v = 0; v < 8; ++v) {
                            int ch = (rb * 2 + rtl) * 16 + hi * 8 + v;
                            out[((size_t)(b * 128 + ch) * 121 + r) * 121 + p] =
                                fmaxf(acc[rtl][c][v], 0.0f);
                        }
                    }
                }
            }
        }
    }
}

extern "C" void kernel_launch(void* const* d_in, const int* in_sizes, int n_in,
                              void* d_out, int out_size, void* d_ws, size_t ws_size,
                              hipStream_t stream) {
    const float* x      = (const float*)d_in[0];
    const float* enc_w  = (const float*)d_in[1];
    const float* enc_b  = (const float*)d_in[2];
    const float* comp_w = (const float*)d_in[3];
    float* out          = (float*)d_out;
    (void)in_sizes; (void)n_in; (void)out_size; (void)d_ws; (void)ws_size;

    dim3 grid(32 * 121);   // one workgroup per (batch, out_row)
    dim3 block(256);       // 8 wave32
    NSM_Conv_10926396801043_kernel<<<grid, block, 0, stream>>>(
        x, enc_w, enc_b, comp_w, out);
}
